// WindowAttentionV2_54924041781539
// MI455X (gfx1250) — matrix operand order repaired and checked by reading, not compiled, not run on hardware
//
#include <hip/hip_runtime.h>

// SwinV2 window attention, fully fused: one workgroup (512 thr = 16 waves) per
// window; wave h owns head h. All intermediates live in LDS; only x is read
// and out is written to HBM (weights stream from L2, pre-converted to bf16
// into d_ws by a tiny prep kernel when ws_size allows).
//
// WMMA fragment layouts used here (bf16 16x16x32; A/B k order is contiguous, not the ISA table order):
//   A (16x32 MxK): lane L, r=L&15, g=L>>4 -> M=r; elements i=0..15 = K(g*16+i)
//                  (one 32B run, loaded as two 16B halves)
//   B (32x16 KxN): lane L -> N=L&15; elements i=0..15 = K(g*16+i) (32B run)
//   C (16x16 f32): vgpr j, lane L -> M = j + 8*(L>>4), N = L&15

#define XS 520                     // LDS row stride (halves) for x / O buffer
#define LOGIT_MAX 4.6051701859880914f   // log(1/0.01)

typedef __attribute__((ext_vector_type(8)))  float   v8f;
typedef __attribute__((ext_vector_type(16))) __bf16  v16bf;
typedef unsigned short u16;
typedef unsigned int   u32;

__device__ __forceinline__ u16 f2bf(float f) {           // f32 -> bf16 (RNE)
  u32 u = __builtin_bit_cast(u32, f);
  return (u16)((u + 0x7FFFu + ((u >> 16) & 1u)) >> 16);
}

union FragU { uint4 q[2]; v16bf v; u16 s[16]; };

__device__ __forceinline__ v16bf load_a_frag(const u16* row, int kbase, int g) {
  FragU u;
  u.q[0] = *(const uint4*)(row + kbase + g * 16);
  u.q[1] = *(const uint4*)(row + kbase + g * 16 + 8);
  return u.v;
}

__device__ __forceinline__ v16bf load_b_frag(const u16* p) {   // 32B contiguous
  FragU u;
  u.q[0] = *(const uint4*)(p);
  u.q[1] = *(const uint4*)(p + 8);
  return u.v;
}

__device__ __forceinline__ v16bf bfrag_f32(const float* p) {   // 16 contig f32
  FragU u;
#pragma unroll
  for (int i = 0; i < 4; ++i) {
    float4 f = ((const float4*)p)[i];
    u.s[4 * i + 0] = f2bf(f.x);
    u.s[4 * i + 1] = f2bf(f.y);
    u.s[4 * i + 2] = f2bf(f.z);
    u.s[4 * i + 3] = f2bf(f.w);
  }
  return u.v;
}

#define WMMA_BF16(A, B, C) \
  __builtin_amdgcn_wmma_f32_16x16x32_bf16(false, (A), false, (B), (short)0, (C), false, false)

// ---- prep: convert w_qkv (786432) + w_proj (262144) f32 -> bf16 into d_ws
__global__ void prep_weights_kernel(const float* __restrict__ wq,
                                    const float* __restrict__ wp,
                                    u16* __restrict__ dst) {
  int i = blockIdx.x * blockDim.x + threadIdx.x;     // float4 units, 262144
  const int NQ = 786432 / 4;
  float4 f = (i < NQ) ? ((const float4*)wq)[i] : ((const float4*)wp)[i - NQ];
  u16* d = dst + i * 4;
  d[0] = f2bf(f.x); d[1] = f2bf(f.y); d[2] = f2bf(f.z); d[3] = f2bf(f.w);
}

template <bool BF16W>
__global__ __launch_bounds__(512, 1)
void winattn_v2_kernel(const float* __restrict__ x,
                       const float* __restrict__ w_qkv,
                       const float* __restrict__ q_bias,
                       const float* __restrict__ v_bias,
                       const float* __restrict__ logit_scale,
                       const float* __restrict__ w_proj,
                       const float* __restrict__ b_proj,
                       const u16* __restrict__ wbf,   // bf16 weights in d_ws
                       float* __restrict__ out) {
  extern __shared__ u16 smem[];
  u16* xs  = smem;                   // [64][XS]  x (bf16); reused as O later
  u16* sqn = xs  + 64 * XS;          // [16][64][32] normalized Q
  u16* skn = sqn + 16 * 64 * 32;     // [16][64][32] normalized K
  u16* svt = skn + 16 * 64 * 32;     // [16][32][64] V transposed (dim-major)
  u16* spb = svt + 16 * 32 * 64;     // [16][16][64] softmax row-block (bf16)

  const int tid  = threadIdx.x;
  const int h    = tid >> 5;         // wave id == head id
  const int lane = tid & 31;
  const int g    = lane >> 4;
  const int ln   = lane & 15;
  const int b    = blockIdx.x;

  // ---- phase 1a: async-stage raw f32 x(window b) into the (not yet used)
  //      qn/kn region (128KB), tracked by ASYNCcnt (CDNA5 async-to-LDS path).
  float* xstage = (float*)sqn;                      // 64*512 f32 = 131072 B
  {
    const float* xg = x + (size_t)b * 64 * 512;
    u32 lds0 = (u32)(uintptr_t)xstage;              // flat low bits = LDS addr
#pragma unroll
    for (int i = 0; i < 16; ++i) {
      int c = tid + i * 512;                        // 16B chunk id, 0..8191
      u32 loff = lds0 + (u32)c * 16u;
      const float* ga = xg + c * 4;
      asm volatile("global_load_async_to_lds_b128 %0, %1, off"
                   :: "v"(loff), "v"(ga) : "memory");
    }
    asm volatile("s_wait_asynccnt 0" ::: "memory");
  }
  __syncthreads();

  // ---- phase 1b: convert staged f32 -> bf16 padded xs
#pragma unroll
  for (int i = 0; i < 16; ++i) {
    int f = tid + i * 512;                          // float4 id, 0..8191
    int row = f >> 7;
    int col = (f & 127) << 2;
    float4 d = ((const float4*)xstage)[f];
    u16* dst = xs + row * XS + col;
    dst[0] = f2bf(d.x); dst[1] = f2bf(d.y);
    dst[2] = f2bf(d.z); dst[3] = f2bf(d.w);
  }
  __syncthreads();

  u16* qn = sqn + h * 64 * 32;
  u16* kn = skn + h * 64 * 32;
  u16* vt = svt + h * 32 * 64;
  u16* pb = spb + h * 16 * 64;

  const v8f vzero = {0.f, 0.f, 0.f, 0.f, 0.f, 0.f, 0.f, 0.f};

  // ---- phase 2: per-head QKV GEMM + bias + cosine norm --------------------
  for (int m = 0; m < 3; ++m) {                     // 0=Q 1=K 2=V
    v8f acc[2][4];
#pragma unroll
    for (int nt = 0; nt < 2; ++nt)
#pragma unroll
      for (int mt = 0; mt < 4; ++mt) acc[nt][mt] = vzero;

    const float* wbase = w_qkv + ((size_t)m * 512 + h * 32) * 512;
    const u16*   wbbase = wbf + ((size_t)m * 512 + h * 32) * 512;
    for (int kc = 0; kc < 16; ++kc) {
      v16bf bfr[2];
#pragma unroll
      for (int nt = 0; nt < 2; ++nt) {              // B[k,n] = w_qkv[ch+n][k]
        if (BF16W)
          bfr[nt] = load_b_frag(wbbase + (size_t)(nt * 16 + ln) * 512 +
                                g * 16 + kc * 32);
        else
          bfr[nt] = bfrag_f32(wbase + (size_t)(nt * 16 + ln) * 512 +
                              g * 16 + kc * 32);
      }
#pragma unroll
      for (int mt = 0; mt < 4; ++mt) {              // A frags shared by both nt
        v16bf af = load_a_frag(xs + (mt * 16 + ln) * XS, kc * 32, g);
        acc[0][mt] = WMMA_BF16(af, bfr[0], acc[0][mt]);
        acc[1][mt] = WMMA_BF16(af, bfr[1], acc[1][mt]);
      }
    }

    if (m == 2) {                                   // V: bias, store transposed
      float vb0 = v_bias[h * 32 + ln];
      float vb1 = v_bias[h * 32 + 16 + ln];
#pragma unroll
      for (int nt = 0; nt < 2; ++nt)
#pragma unroll
        for (int mt = 0; mt < 4; ++mt)
#pragma unroll
          for (int j = 0; j < 8; ++j)
            vt[(nt * 16 + ln) * 64 + mt * 16 + j + g * 8] =
                f2bf(acc[nt][mt][j] + (nt ? vb1 : vb0));
    } else {
      if (m == 0) {                                 // Q bias before norm
        float qb0 = q_bias[h * 32 + ln];
        float qb1 = q_bias[h * 32 + 16 + ln];
#pragma unroll
        for (int mt = 0; mt < 4; ++mt)
#pragma unroll
          for (int j = 0; j < 8; ++j) {
            acc[0][mt][j] += qb0;
            acc[1][mt][j] += qb1;
          }
      }
      u16* dstm = (m == 0) ? qn : kn;
#pragma unroll
      for (int mt = 0; mt < 4; ++mt)
#pragma unroll
        for (int j = 0; j < 8; ++j) {               // row L2-norm over d=32
          float a0 = acc[0][mt][j], a1 = acc[1][mt][j];
          float ss = a0 * a0 + a1 * a1;
          ss += __shfl_xor(ss, 1, 32);
          ss += __shfl_xor(ss, 2, 32);
          ss += __shfl_xor(ss, 4, 32);
          ss += __shfl_xor(ss, 8, 32);              // stays in 16-lane group
          float inv = 1.0f / fmaxf(sqrtf(ss), 1e-12f);
          u16* dst = dstm + (mt * 16 + j + g * 8) * 32;
          dst[ln]      = f2bf(a0 * inv);
          dst[16 + ln] = f2bf(a1 * inv);
        }
    }
  }
  __syncthreads();   // all waves done reading xs -> safe to overwrite with O

  // ---- phase 3: attention (logits -> softmax -> P.V) ----------------------
  float scale = __expf(fminf(logit_scale[h], LOGIT_MAX));
  for (int mt = 0; mt < 4; ++mt) {
    v16bf aq = load_a_frag(qn + (mt * 16 + ln) * 32, 0, g);
    v8f p[4];
#pragma unroll
    for (int nt = 0; nt < 4; ++nt) {                // B = kn^T: contiguous rows
      v16bf bk = load_b_frag(kn + (nt * 16 + ln) * 32 + g * 16);
      p[nt] = WMMA_BF16(aq, bk, vzero);             // K=32 = D, single WMMA
    }
#pragma unroll
    for (int j = 0; j < 8; ++j) {                   // softmax over 64 cols
      float e0 = p[0][j] * scale, e1 = p[1][j] * scale;
      float e2 = p[2][j] * scale, e3 = p[3][j] * scale;
      float mx = fmaxf(fmaxf(e0, e1), fmaxf(e2, e3));
      mx = fmaxf(mx, __shfl_xor(mx, 1, 32));
      mx = fmaxf(mx, __shfl_xor(mx, 2, 32));
      mx = fmaxf(mx, __shfl_xor(mx, 4, 32));
      mx = fmaxf(mx, __shfl_xor(mx, 8, 32));
      e0 = __expf(e0 - mx); e1 = __expf(e1 - mx);
      e2 = __expf(e2 - mx); e3 = __expf(e3 - mx);
      float s = e0 + e1 + e2 + e3;
      s += __shfl_xor(s, 1, 32);
      s += __shfl_xor(s, 2, 32);
      s += __shfl_xor(s, 4, 32);
      s += __shfl_xor(s, 8, 32);
      float inv = 1.0f / s;
      u16* pr = pb + (j + g * 8) * 64;
      pr[ln]      = f2bf(e0 * inv);
      pr[16 + ln] = f2bf(e1 * inv);
      pr[32 + ln] = f2bf(e2 * inv);
      pr[48 + ln] = f2bf(e3 * inv);
    }
    v8f o0 = vzero, o1 = vzero;                     // out_block[16x32] = P.V
#pragma unroll
    for (int kc = 0; kc < 2; ++kc) {
      v16bf ap  = load_a_frag(pb + ln * 64, kc * 32, g);
      v16bf bv0 = load_b_frag(vt + ln * 64 + kc * 32 + g * 16);
      v16bf bv1 = load_b_frag(vt + (16 + ln) * 64 + kc * 32 + g * 16);
      o0 = WMMA_BF16(ap, bv0, o0);
      o1 = WMMA_BF16(ap, bv1, o1);
    }
#pragma unroll
    for (int j = 0; j < 8; ++j) {                   // write O into reused xs
      u16* orow = xs + (mt * 16 + j + g * 8) * XS + h * 32;
      orow[ln]      = f2bf(o0[j]);
      orow[16 + ln] = f2bf(o1[j]);
    }
  }
  __syncthreads();   // O[64x512] complete across all heads

  // ---- phase 4: output projection + bias -> d_out -------------------------
  {
    v8f acc[2][4];
#pragma unroll
    for (int nt = 0; nt < 2; ++nt)
#pragma unroll
      for (int mt = 0; mt < 4; ++mt) acc[nt][mt] = vzero;

    const u16* wpb = wbf + 786432;                  // bf16 w_proj base
    for (int kc = 0; kc < 16; ++kc) {
      v16bf bfr[2];
#pragma unroll
      for (int nt = 0; nt < 2; ++nt) {
        if (BF16W)
          bfr[nt] = load_b_frag(wpb + (size_t)(h * 32 + nt * 16 + ln) * 512 +
                                g * 16 + kc * 32);
        else
          bfr[nt] = bfrag_f32(w_proj + (size_t)(h * 32 + nt * 16 + ln) * 512 +
                              g * 16 + kc * 32);
      }
#pragma unroll
      for (int mt = 0; mt < 4; ++mt) {
        v16bf af = load_a_frag(xs + (mt * 16 + ln) * XS, kc * 32, g);
        acc[0][mt] = WMMA_BF16(af, bfr[0], acc[0][mt]);
        acc[1][mt] = WMMA_BF16(af, bfr[1], acc[1][mt]);
      }
    }
#pragma unroll
    for (int nt = 0; nt < 2; ++nt) {
      float bias = b_proj[h * 32 + nt * 16 + ln];
#pragma unroll
      for (int mt = 0; mt < 4; ++mt)
#pragma unroll
        for (int j = 0; j < 8; ++j) {
          int row = mt * 16 + j + g * 8;
          out[((size_t)b * 64 + row) * 512 + h * 32 + nt * 16 + ln] =
              acc[nt][mt][j] + bias;
        }
    }
  }
}

extern "C" void kernel_launch(void* const* d_in, const int* in_sizes, int n_in,
                              void* d_out, int out_size, void* d_ws, size_t ws_size,
                              hipStream_t stream) {
  (void)in_sizes; (void)n_in; (void)out_size;
  const float* x  = (const float*)d_in[0];
  const float* wq = (const float*)d_in[1];
  const float* qb = (const float*)d_in[2];
  const float* vb = (const float*)d_in[3];
  const float* ls = (const float*)d_in[4];
  const float* wp = (const float*)d_in[5];
  const float* bp = (const float*)d_in[6];
  float* o = (float*)d_out;
  const size_t shmem =
      (size_t)(64 * XS + 3 * 16 * 64 * 32 + 16 * 16 * 64) * sizeof(u16); // 295936 B
  const size_t W_BYTES = (size_t)(786432 + 262144) * sizeof(u16);       // 2 MB

  if (ws_size >= W_BYTES) {
    u16* wbf = (u16*)d_ws;
    prep_weights_kernel<<<dim3(1024), dim3(256), 0, stream>>>(wq, wp, wbf);
    winattn_v2_kernel<true><<<dim3(2048), dim3(512), shmem, stream>>>(
        x, wq, qb, vb, ls, wp, bp, wbf, o);
  } else {
    winattn_v2_kernel<false><<<dim3(2048), dim3(512), shmem, stream>>>(
        x, wq, qb, vb, ls, wp, bp, (const u16*)nullptr, o);
  }
}
